// Attention_14465449853282
// MI455X (gfx1250) — 16-bit and 8-bit matrix operand order repaired and checked by reading, compiled, not run on hardware
//
#include <hip/hip_runtime.h>
#include <hip/hip_bf16.h>
#include <stdint.h>

typedef __attribute__((ext_vector_type(16))) _Float16 v16h;
typedef __attribute__((ext_vector_type(8)))  _Float16 v8h;
typedef __attribute__((ext_vector_type(8)))  float    v8f;
typedef __attribute__((ext_vector_type(8)))  int      v8i;

constexpr int B_ = 8, N_ = 2048, C_ = 384, H_ = 6, D_ = 64;
constexpr int M_ = B_ * N_;
constexpr int C3 = 3 * C_;

__global__ __launch_bounds__(256) void k_reduce_abs_pass1(const float* __restrict__ w, int n,
                                                          float* __restrict__ partial) {
  __shared__ float red[256];
  int t = threadIdx.x;
  float s = 0.f;
  for (int i = blockIdx.x * 256 + t; i < n; i += gridDim.x * 256) s += fabsf(w[i]);
  red[t] = s; __syncthreads();
  for (int off = 128; off > 0; off >>= 1) {
    if (t < off) red[t] += red[t + off];
    __syncthreads();
  }
  if (t == 0) partial[blockIdx.x] = red[0];
}

__global__ __launch_bounds__(256) void k_reduce_pass2(const float* __restrict__ partial, int g,
                                                      float* __restrict__ sum_out) {
  __shared__ float red[256];
  int t = threadIdx.x;
  float s = 0.f;
  for (int i = t; i < g; i += 256) s += partial[i];
  red[t] = s; __syncthreads();
  for (int off = 128; off > 0; off >>= 1) {
    if (t < off) red[t] += red[t + off];
    __syncthreads();
  }
  if (t == 0) sum_out[0] = red[0];
}

__global__ __launch_bounds__(256) void k_quant_w(const float* __restrict__ w, int8_t* __restrict__ wq,
                                                 int n, const float* __restrict__ sum, float invcnt) {
  int i = blockIdx.x * 256 + threadIdx.x;
  if (i >= n) return;
  float mean = fmaxf(sum[0] * invcnt, 1e-5f);
  float sc = 1.f / mean;
  float q = fminf(fmaxf(rintf(w[i] * sc), -1.f), 1.f);
  wq[i] = (int8_t)q;
}

__global__ __launch_bounds__(128) void k_quant_act(const float* __restrict__ x, int8_t* __restrict__ xq,
                                                   float* __restrict__ rec) {
  __shared__ float red[128];
  int row = blockIdx.x, t = threadIdx.x;
  const float* xr = x + (size_t)row * C_;
  float v0 = xr[t], v1 = xr[t + 128], v2 = xr[t + 256];
  float am = fmaxf(fabsf(v0), fmaxf(fabsf(v1), fabsf(v2)));
  red[t] = am; __syncthreads();
  for (int off = 64; off > 0; off >>= 1) {
    if (t < off) red[t] = fmaxf(red[t], red[t + off]);
    __syncthreads();
  }
  float mx = fmaxf(red[0], 1e-5f);
  float sc = 128.f / mx;
  int8_t* o = xq + (size_t)row * C_;
  o[t]       = (int8_t)fminf(fmaxf(rintf(v0 * sc), -128.f), 127.f);
  o[t + 128] = (int8_t)fminf(fmaxf(rintf(v1 * sc), -128.f), 127.f);
  o[t + 256] = (int8_t)fminf(fmaxf(rintf(v2 * sc), -128.f), 127.f);
  if (t == 0) rec[row] = mx * (1.f / 128.f);
}

static __device__ __forceinline__ v8i gemm_tile_iu8(const int8_t* __restrict__ xq,
                                                    const int8_t* __restrict__ wq,
                                                    int tm, int tn, int lrow, int hi) {
  v8i acc = {};
  const int8_t* arow = xq + (size_t)(tm + lrow) * C_;
  const int8_t* brow = wq + (size_t)(tn + lrow) * C_;
#pragma unroll
  for (int kb = 0; kb < C_; kb += 64) {
    __builtin_prefetch(arow + kb + 64, 0, 3);
    v8i a, b;
#pragma unroll
    for (int j = 0; j < 8; ++j) {
      int ka = kb + (j & 3) * 4 + (j >> 2) * 32 + hi * 16;
      int kw = kb + (j & 3) * 4 + (j >> 2) * 32 + hi * 16;
      a[j] = *(const int*)(arow + ka);
      b[j] = *(const int*)(brow + kw);
    }
    acc = __builtin_amdgcn_wmma_i32_16x16x64_iu8(true, a, true, b, acc, false, false);
  }
  return acc;
}

__global__ __launch_bounds__(128) void k_gemm_qkv(const int8_t* __restrict__ xq,
                                                  const float* __restrict__ arecip,
                                                  const int8_t* __restrict__ wq,
                                                  const float* __restrict__ wsum,
                                                  const float* __restrict__ bias,
                                                  _Float16* __restrict__ q16,
                                                  _Float16* __restrict__ k16,
                                                  _Float16* __restrict__ vt16) {
  int tid = threadIdx.x, lane = tid & 31, lrow = lane & 15, hi = lane >> 4;
  int tile = blockIdx.x * 4 + (tid >> 5);
  const int NT = C3 / 16;
  int tm = (tile / NT) * 16, tn = (tile % NT) * 16;
  v8i acc = gemm_tile_iu8(xq, wq, tm, tn, lrow, hi);
  float wdq = fmaxf(wsum[0] * (1.f / (float)(C3 * C_)), 1e-5f);
  int col = tn + lrow;
  float bcol = bias[col];
  int sec = col / C_, wc = col % C_;
  int h = wc >> 6, d = wc & 63;
#pragma unroll
  for (int r = 0; r < 8; ++r) {
    int m = tm + r + hi * 8;
    float v = (float)acc[r] * arecip[m] * wdq + bcol;
    int b = m >> 11, nn = m & 2047;
    _Float16 hv = (_Float16)v;
    size_t bh = (size_t)(b * H_ + h);
    if (sec == 0)      q16[(bh * N_ + nn) * D_ + d] = hv;
    else if (sec == 1) k16[(bh * N_ + nn) * D_ + d] = hv;
    else               vt16[(bh * D_ + d) * N_ + nn] = hv;
  }
}

__global__ __launch_bounds__(128) void k_gemm_proj(const int8_t* __restrict__ xq,
                                                   const float* __restrict__ arecip,
                                                   const int8_t* __restrict__ wq,
                                                   const float* __restrict__ wsum,
                                                   const float* __restrict__ bias,
                                                   float* __restrict__ out) {
  int tid = threadIdx.x, lane = tid & 31, lrow = lane & 15, hi = lane >> 4;
  int tile = blockIdx.x * 4 + (tid >> 5);
  const int NT = C_ / 16;
  int tm = (tile / NT) * 16, tn = (tile % NT) * 16;
  v8i acc = gemm_tile_iu8(xq, wq, tm, tn, lrow, hi);
  float wdq = fmaxf(wsum[0] * (1.f / (float)(C_ * C_)), 1e-5f);
  int col = tn + lrow;
  float bcol = bias[col];
#pragma unroll
  for (int r = 0; r < 8; ++r) {
    int m = tm + r + hi * 8;
    out[(size_t)m * C_ + col] = (float)acc[r] * arecip[m] * wdq + bcol;
  }
}

union U16h { v16h h; v8h h8[2]; };

__global__ __launch_bounds__(128) void k_attn(const _Float16* __restrict__ q16,
                                              const _Float16* __restrict__ k16,
                                              const _Float16* __restrict__ vt16,
                                              float* __restrict__ out) {
  int tid = threadIdx.x, lane = tid & 31, lrow = lane & 15, hi = lane >> 4;
  int wid = blockIdx.x * 4 + (tid >> 5);
  int qt = wid & 127;
  int bh = wid >> 7;
  int qbase = qt * 16;

  U16h qb[2];
  const _Float16* qrow = q16 + ((size_t)bh * N_ + qbase + lrow) * D_;
#pragma unroll
  for (int f = 0; f < 2; ++f) qb[f].h = *(const v16h*)(qrow + f * 32 + hi * 16);

  v8f o[4] = {v8f{}, v8f{}, v8f{}, v8f{}};
  float m_run = -3.0e38f, l_run = 0.f;
  const float sm_scale = 0.125f;

  for (int kb = 0; kb < N_; kb += 32) {
    U16h ka[2][2];
#pragma unroll
    for (int kt = 0; kt < 2; ++kt) {
      const _Float16* krow = k16 + ((size_t)bh * N_ + kb + kt * 16 + lrow) * D_;
#pragma unroll
      for (int f = 0; f < 2; ++f) {
        const _Float16* p = krow + f * 32 + hi * 16;
        ka[kt][f].h8[0] = *(const v8h*)(p);
        ka[kt][f].h8[1] = *(const v8h*)(p + 8);
      }
    }
    v8f s0 = {}, s1 = {};
    s0 = __builtin_amdgcn_wmma_f32_16x16x32_f16(false, ka[0][0].h, false, qb[0].h, (short)0, s0, false, false);
    s0 = __builtin_amdgcn_wmma_f32_16x16x32_f16(false, ka[0][1].h, false, qb[1].h, (short)0, s0, false, false);
    s1 = __builtin_amdgcn_wmma_f32_16x16x32_f16(false, ka[1][0].h, false, qb[0].h, (short)0, s1, false, false);
    s1 = __builtin_amdgcn_wmma_f32_16x16x32_f16(false, ka[1][1].h, false, qb[1].h, (short)0, s1, false, false);

    float mt = -3.0e38f;
#pragma unroll
    for (int r = 0; r < 8; ++r) {
      s0[r] *= sm_scale; s1[r] *= sm_scale;
      mt = fmaxf(mt, fmaxf(s0[r], s1[r]));
    }
    mt = fmaxf(mt, __shfl_xor(mt, 16, 32));
    float mnew = fmaxf(m_run, mt);
    float corr = __expf(m_run - mnew);
    float p0[8], p1[8], lp = 0.f;
#pragma unroll
    for (int r = 0; r < 8; ++r) {
      p0[r] = __expf(s0[r] - mnew);
      p1[r] = __expf(s1[r] - mnew);
      lp += p0[r] + p1[r];
    }
    lp += __shfl_xor(lp, 16, 32);
    l_run = l_run * corr + lp;
    m_run = mnew;
#pragma unroll
    for (int t = 0; t < 4; ++t)
#pragma unroll
      for (int r = 0; r < 8; ++r) o[t][r] *= corr;

    U16h pb;
#pragma unroll
    for (int r = 0; r < 8; ++r) {
      float x0 = __shfl_xor(p0[r], 16, 32);
      float x1 = __shfl_xor(p1[r], 16, 32);
      float lo = hi ? x1 : p0[r];
      float hv = hi ? p1[r] : x0;
      pb.h[r]     = (_Float16)lo;
      pb.h[r + 8] = (_Float16)hv;
    }

#pragma unroll
    for (int t = 0; t < 4; ++t) {
      U16h va;
      const _Float16* vrow = vt16 + ((size_t)bh * D_ + t * 16 + lrow) * N_ + kb + hi * 16;
      va.h8[0] = *(const v8h*)(vrow);
      va.h8[1] = *(const v8h*)(vrow + 8);
      o[t] = __builtin_amdgcn_wmma_f32_16x16x32_f16(false, va.h, false, pb.h, (short)0, o[t], false, false);
    }
  }

  float rl = 1.f / l_run;
  int b = bh / H_, h = bh % H_;
  float* orow = out + ((size_t)(b * N_ + qbase + lrow)) * C_ + h * 64;
#pragma unroll
  for (int t = 0; t < 4; ++t)
#pragma unroll
    for (int r = 0; r < 8; ++r)
      orow[t * 16 + r + hi * 8] = o[t][r] * rl;
}

extern "C" void kernel_launch(void* const* d_in, const int* in_sizes, int n_in,
                              void* d_out, int out_size, void* d_ws, size_t ws_size,
                              hipStream_t stream) {
  (void)in_sizes; (void)n_in; (void)out_size; (void)ws_size;
  const float* x      = (const float*)d_in[0];
  const float* qkv_w  = (const float*)d_in[1];
  const float* qkv_b  = (const float*)d_in[2];
  const float* proj_w = (const float*)d_in[3];
  const float* proj_b = (const float*)d_in[4];
  float* out = (float*)d_out;

  size_t off = 0;
  auto alloc = [&](size_t bytes) -> void* {
    void* p = (char*)d_ws + off;
    off = (off + bytes + 255) & ~(size_t)255;
    return p;
  };
  float*    scal    = (float*)alloc(2 * sizeof(float));
  float*    part0   = (float*)alloc(256 * sizeof(float));
  float*    part1   = (float*)alloc(128 * sizeof(float));
  int8_t*   xq      = (int8_t*)alloc((size_t)M_ * C_);
  float*    arecip  = (float*)alloc((size_t)M_ * sizeof(float));
  int8_t*   wq_qkv  = (int8_t*)alloc((size_t)C3 * C_);
  int8_t*   wq_proj = (int8_t*)alloc((size_t)C_ * C_);
  _Float16* q16     = (_Float16*)alloc((size_t)B_ * H_ * N_ * D_ * 2);
  _Float16* k16     = (_Float16*)alloc((size_t)B_ * H_ * N_ * D_ * 2);
  _Float16* vt16    = (_Float16*)alloc((size_t)B_ * H_ * N_ * D_ * 2);
  float*    attn    = (float*)alloc((size_t)M_ * C_ * sizeof(float));
  int8_t*   xq2     = (int8_t*)alloc((size_t)M_ * C_);
  float*    arecip2 = (float*)alloc((size_t)M_ * sizeof(float));

  k_reduce_abs_pass1<<<256, 256, 0, stream>>>(qkv_w, C3 * C_, part0);
  k_reduce_pass2<<<1, 256, 0, stream>>>(part0, 256, scal + 0);
  k_reduce_abs_pass1<<<128, 256, 0, stream>>>(proj_w, C_ * C_, part1);
  k_reduce_pass2<<<1, 256, 0, stream>>>(part1, 128, scal + 1);

  k_quant_w<<<(C3 * C_ + 255) / 256, 256, 0, stream>>>(qkv_w, wq_qkv, C3 * C_, scal + 0,
                                                       1.f / (float)(C3 * C_));
  k_quant_w<<<(C_ * C_ + 255) / 256, 256, 0, stream>>>(proj_w, wq_proj, C_ * C_, scal + 1,
                                                       1.f / (float)(C_ * C_));

  k_quant_act<<<M_, 128, 0, stream>>>(x, xq, arecip);
  k_gemm_qkv<<<(M_ / 16) * (C3 / 16) / 4, 128, 0, stream>>>(xq, arecip, wq_qkv, scal + 0,
                                                            qkv_b, q16, k16, vt16);

  k_attn<<<B_ * H_ * (N_ / 16) / 4, 128, 0, stream>>>(q16, k16, vt16, attn);

  k_quant_act<<<M_, 128, 0, stream>>>(attn, xq2, arecip2);
  k_gemm_proj<<<(M_ / 16) * (C_ / 16) / 4, 128, 0, stream>>>(xq2, arecip2, wq_proj, scal + 1,
                                                             proj_b, out);
}
